// Transducer_55594056679538
// MI455X (gfx1250) — matrix operand order repaired and checked by reading, not compiled, not run on hardware
//
#include <hip/hip_runtime.h>
#include <hip/hip_bf16.h>

// ---------------------------------------------------------------------------
// RNN-T (Transducer) forward loss on MI455X (gfx1250, wave32, WMMA bf16)
//
// Phases:
//   1) pack_w          : f32 weights -> bf16 WMMA B-fragment layout (7x)
//   2) gemm_bf16       : X0 = xs @ Wih0^T + b            [8000,1024]
//   3) lstm_seq        : encoder layer0 recurrence       (1 WG, 16 waves)
//   4) gemm_bf16       : X1 = h0 @ Wih1^T + b (reuse X0 buffer)
//   5) lstm_seq        : encoder layer1 recurrence
//   6) gemm_bf16       : enc logits = h1 @ enc_W^T + b   [8000,128]
//   7) dec_gather      : Xd from one-hot embedding (column gather of d_Wih)
//   8) lstm_seq        : decoder recurrence (51 steps)
//   9) gemm_bf16       : dec logits = hd @ dec_W^T + b   [816,128]
//  10) joint_lse       : per-(b,t) wave: logsumexp over V=128, emit blank/lab
//  11) alpha_loss      : 16 serial alpha recursions + reduce -> scalar loss
// ---------------------------------------------------------------------------

#define BB   16
#define TT   500
#define UU   50
#define U1   51
#define VV   128
#define INP  80
#define HH   256
#define GG   1024   // 4*H

typedef __attribute__((ext_vector_type(16))) __bf16 v16bf;
typedef __attribute__((ext_vector_type(8)))  float  v8f;

// Position of element (m,k) inside packed WMMA A-fragment storage:
// per 16x32 k-tile: 32 lanes * 16 halves, laid out so a lane loads one v16bf.
__device__ __forceinline__ int apack_pos(int m, int k) {
  int kt   = k >> 5;
  int kk   = k & 31;
  int half = (kk >> 3) & 1;
  int e    = ((kk >> 4) << 3) | (kk & 7);
  int ln   = (half << 4) | m;
  return kt * 512 + ln * 16 + e;
}

__device__ __forceinline__ float sigf(float x) {
  return 1.0f / (1.0f + __expf(-x));
}

// ---------------------------------------------------------------------------
// Pack W [N][K] (f32, row-major) into bf16 WMMA B-fragment tiles for B = W^T.
// Tile (kt,nt): 32 lanes x 16 bf16 each; lane l holds col n = nt*16+(l&15),
// K rows kt*32 + ((e>>3)<<4) + ((l>>4)<<3) + (e&7). Zero-pads K to a multiple of 32.
// ---------------------------------------------------------------------------
__global__ __launch_bounds__(256) void pack_w(const float* __restrict__ W,
                                              __bf16* __restrict__ pack,
                                              int N, int K) {
  int ktiles = (K + 31) >> 5;
  int ntiles = N >> 4;
  long total = (long)ktiles * ntiles * 512;
  long idx = (long)blockIdx.x * 256 + threadIdx.x;
  if (idx >= total) return;
  int e    = (int)(idx & 15);
  int lane = (int)((idx >> 4) & 31);
  long tile = idx >> 9;
  int nt = (int)(tile % ntiles);
  int kt = (int)(tile / ntiles);
  int n  = nt * 16 + (lane & 15);
  int k  = kt * 32 + ((e >> 3) << 4) + (((lane >> 4) & 1) << 3) + (e & 7);
  float v = (k < K) ? W[(long)n * K + k] : 0.0f;
  pack[idx] = (__bf16)v;
}

// ---------------------------------------------------------------------------
// C[M,N] = A[M,K](f32->bf16) @ Bpack + bias0 + bias1  (f32 accumulate)
// grid = (N/128, M/16), 256 threads = 8 waves, one 16x16 tile per wave.
// A tile (16x32 f32) staged into LDS in packed fragment layout by 128 threads:
// one global_load_b128 (4 f32) -> cvt_pk_bf16 -> one ds_store_b64 each.
// The k-loop is split into a branch-free full-tile loop plus one peeled,
// per-element-guarded tail tile (only taken when K % 32 != 0, i.e. K=80).
// ---------------------------------------------------------------------------
__global__ __launch_bounds__(256) void gemm_bf16(const float* __restrict__ A,
                                                 const __bf16* __restrict__ Bpack,
                                                 const float* __restrict__ bias0,
                                                 const float* __restrict__ bias1,
                                                 float* __restrict__ C,
                                                 int M, int N, int K) {
  __shared__ __bf16 aLds[512];
  int lane = threadIdx.x & 31;
  int wave = threadIdx.x >> 5;
  int m0 = blockIdx.y << 4;
  int nt = (blockIdx.x << 3) + wave;
  int ntiles = N >> 4;
  int ktfull = K >> 5;               // fully-populated 32-wide k tiles
  v8f acc = {0.f, 0.f, 0.f, 0.f, 0.f, 0.f, 0.f, 0.f};

  int sm = threadIdx.x >> 3;         // staging row   0..15 (threads 0..127)
  int sk = (threadIdx.x & 7) << 2;   // staging k*4   0..28
  __bf16* sd = aLds + apack_pos(sm, sk);  // 4 consecutive slots, 8B aligned
  const float* arow = A + (size_t)(m0 + sm) * K;

  // ---- branch-free main loop ----
  for (int kt = 0; kt < ktfull; ++kt) {
    __syncthreads();
    if (threadIdx.x < 128) {
      float4 v = *reinterpret_cast<const float4*>(arow + (kt << 5) + sk);
      sd[0] = (__bf16)v.x;
      sd[1] = (__bf16)v.y;
      sd[2] = (__bf16)v.z;
      sd[3] = (__bf16)v.w;
    }
    __syncthreads();
    v16bf a = *(const v16bf*)(aLds + lane * 16);
    v16bf b = *(const v16bf*)(Bpack + ((size_t)(kt * ntiles + nt) * 32 + lane) * 16);
    acc = __builtin_amdgcn_wmma_f32_16x16x32_bf16(false, a, false, b,
                                                  (short)0, acc, false, false);
  }

  // ---- peeled K-tail tile (K % 32 != 0) ----
  if (K & 31) {
    int kt = ktfull;
    __syncthreads();
    if (threadIdx.x < 128) {
      int k = (kt << 5) + sk;
      const float* ap = arow + k;
      float x0 = (k     < K) ? ap[0] : 0.0f;
      float x1 = (k + 1 < K) ? ap[1] : 0.0f;
      float x2 = (k + 2 < K) ? ap[2] : 0.0f;
      float x3 = (k + 3 < K) ? ap[3] : 0.0f;
      sd[0] = (__bf16)x0;
      sd[1] = (__bf16)x1;
      sd[2] = (__bf16)x2;
      sd[3] = (__bf16)x3;
    }
    __syncthreads();
    v16bf a = *(const v16bf*)(aLds + lane * 16);
    v16bf b = *(const v16bf*)(Bpack + ((size_t)(kt * ntiles + nt) * 32 + lane) * 16);
    acc = __builtin_amdgcn_wmma_f32_16x16x32_bf16(false, a, false, b,
                                                  (short)0, acc, false, false);
  }

  int col = (nt << 4) + (lane & 15);
  float bv = (bias0 ? bias0[col] : 0.0f) + (bias1 ? bias1[col] : 0.0f);
  int rbase = m0 + ((lane >> 4) << 3);
#pragma unroll
  for (int r = 0; r < 8; ++r)
    C[(size_t)(rbase + r) * N + col] = acc[r] + bv;
}

// ---------------------------------------------------------------------------
// Sequential LSTM recurrence, B=16 (one WMMA M-tile), H=256, gates=1024.
// One workgroup, 512 threads = 16 waves. Wave w owns hidden units
// j in [w*16, w*16+16) -> gate tiles {w, w+16, w+32, w+48} = (i,f,g,o) for
// the SAME units, so c-update stays in the wave's accumulator registers.
// h (bf16, WMMA-A-packed) lives in 8KB LDS; 2 barriers per step.
// X: [B*Tsteps, 1024] precomputed input-side gates (+biases).
// ---------------------------------------------------------------------------
__global__ __launch_bounds__(512) void lstm_seq(const float* __restrict__ X,
                                                const __bf16* __restrict__ Wpack,
                                                float* __restrict__ hseq,
                                                int Tsteps) {
  __shared__ __bf16 hp[8 * 512];  // packed A fragments for kt = 0..7
  int lane = threadIdx.x & 31;
  int wave = threadIdx.x >> 5;  // 0..15

  for (int i = threadIdx.x; i < 8 * 512; i += 512) hp[i] = (__bf16)0.0f;
  __syncthreads();

  float c[8];
#pragma unroll
  for (int r = 0; r < 8; ++r) c[r] = 0.0f;

  int mrow = (lane >> 4) << 3;    // batch-row base (0 or 8)
  int j    = wave * 16 + (lane & 15);  // hidden unit column

  for (int t = 0; t < Tsteps; ++t) {
    v8f a0 = {0.f,0.f,0.f,0.f,0.f,0.f,0.f,0.f};
    v8f a1 = a0, a2 = a0, a3 = a0;
#pragma unroll
    for (int kt = 0; kt < 8; ++kt) {
      v16bf a = *(const v16bf*)(hp + kt * 512 + lane * 16);
      v16bf b0 = *(const v16bf*)(Wpack + ((size_t)(kt * 64 + ( 0 + wave)) * 32 + lane) * 16);
      v16bf b1 = *(const v16bf*)(Wpack + ((size_t)(kt * 64 + (16 + wave)) * 32 + lane) * 16);
      v16bf b2 = *(const v16bf*)(Wpack + ((size_t)(kt * 64 + (32 + wave)) * 32 + lane) * 16);
      v16bf b3 = *(const v16bf*)(Wpack + ((size_t)(kt * 64 + (48 + wave)) * 32 + lane) * 16);
      a0 = __builtin_amdgcn_wmma_f32_16x16x32_bf16(false, a, false, b0, (short)0, a0, false, false);
      a1 = __builtin_amdgcn_wmma_f32_16x16x32_bf16(false, a, false, b1, (short)0, a1, false, false);
      a2 = __builtin_amdgcn_wmma_f32_16x16x32_bf16(false, a, false, b2, (short)0, a2, false, false);
      a3 = __builtin_amdgcn_wmma_f32_16x16x32_bf16(false, a, false, b3, (short)0, a3, false, false);
    }
    __syncthreads();  // all waves done reading hp (previous h)

#pragma unroll
    for (int r = 0; r < 8; ++r) {
      int b = mrow + r;
      const float* xr = X + ((size_t)b * Tsteps + t) * GG;
      float gi = a0[r] + xr[0 * HH + j];
      float gf = a1[r] + xr[1 * HH + j];
      float gg = a2[r] + xr[2 * HH + j];
      float go = a3[r] + xr[3 * HH + j];
      float cc = sigf(gf) * c[r] + sigf(gi) * tanhf(gg);
      float hh = sigf(go) * tanhf(cc);
      c[r] = cc;
      hp[apack_pos(b, j)] = (__bf16)hh;
      hseq[((size_t)b * Tsteps + t) * HH + j] = hh;
    }
    __syncthreads();  // h published for next step
  }
}

// ---------------------------------------------------------------------------
// Decoder input gates from one-hot embedding: Xd[b,u,:] = bih+bhh (+ column
// y-1 of d_Wih if label y>0). d_Wih is [1024, V-1].
// ---------------------------------------------------------------------------
__global__ __launch_bounds__(256) void dec_gather(const int* __restrict__ ymat,
                                                  const float* __restrict__ dWih,
                                                  const float* __restrict__ bih,
                                                  const float* __restrict__ bhh,
                                                  float* __restrict__ Xd) {
  int idx = blockIdx.x * 256 + threadIdx.x;
  if (idx >= BB * U1 * GG) return;
  int n   = idx & (GG - 1);
  int row = idx >> 10;  // b*U1 + u
  int y = ymat[row];
  float v = bih[n] + bhh[n];
  if (y > 0) v += dWih[(size_t)n * (VV - 1) + (y - 1)];
  Xd[idx] = v;
}

// ---------------------------------------------------------------------------
// Joint + log-softmax, never materializing lp[B,T,U+1,V]:
// one wave per (b,t); enc row (128 f32) in registers (4/lane); stream the 51
// dec rows (L2-resident); wave32 shuffle max/sum-exp; emit blank/label lp.
// grid = B*T/8 blocks of 256 threads (8 waves).
// ---------------------------------------------------------------------------
__global__ __launch_bounds__(256) void joint_lse(const float* __restrict__ enc,
                                                 const float* __restrict__ dec,
                                                 const int* __restrict__ ys,
                                                 float* __restrict__ blank,
                                                 float* __restrict__ lab) {
  int lane = threadIdx.x & 31;
  int wave = threadIdx.x >> 5;
  int bt = blockIdx.x * 8 + wave;  // b*T + t
  int b = bt / TT;
  const float* er = enc + (size_t)bt * VV;
  float e0 = er[lane], e1 = er[lane + 32], e2 = er[lane + 64], e3 = er[lane + 96];

  for (int u = 0; u < U1; ++u) {
    const float* dr = dec + ((size_t)b * U1 + u) * VV;
    float s0 = e0 + dr[lane];
    float s1 = e1 + dr[lane + 32];
    float s2 = e2 + dr[lane + 64];
    float s3 = e3 + dr[lane + 96];
    float m = fmaxf(fmaxf(s0, s1), fmaxf(s2, s3));
#pragma unroll
    for (int off = 16; off >= 1; off >>= 1) m = fmaxf(m, __shfl_xor(m, off, 32));
    float sum = __expf(s0 - m) + __expf(s1 - m) + __expf(s2 - m) + __expf(s3 - m);
#pragma unroll
    for (int off = 16; off >= 1; off >>= 1) sum += __shfl_xor(sum, off, 32);
    float lse = m + __logf(sum);

    float sb = __shfl(s0, 0, 32);  // v = BLANK = 0
    blank[(size_t)bt * U1 + u] = sb - lse;

    if (u < UU) {
      int y  = ys[b * UU + u];
      int yl = y & 31;
      int ye = y >> 5;
      float cand = (ye == 0) ? s0 : (ye == 1) ? s1 : (ye == 2) ? s2 : s3;
      float pick = (lane == yl) ? cand : 0.0f;
#pragma unroll
      for (int off = 16; off >= 1; off >>= 1) pick += __shfl_xor(pick, off, 32);
      lab[(size_t)bt * UU + u] = pick - lse;
    }
  }
}

// ---------------------------------------------------------------------------
// Alpha recursion per batch element (serial in t,u) + final loss reduce.
// One block of 32 threads, lanes 0..15 active.
// ---------------------------------------------------------------------------
__device__ __forceinline__ float logaddexpf_(float a, float b) {
  float mx = fmaxf(a, b);
  float mn = fminf(a, b);
  if (mx == -INFINITY) return mx;
  return mx + log1pf(__expf(mn - mx));
}

__global__ void alpha_loss(const float* __restrict__ blank,
                           const float* __restrict__ lab,
                           const int* __restrict__ xlen,
                           const int* __restrict__ ylen,
                           float* __restrict__ out) {
  int b = threadIdx.x;
  float ll = 0.0f;
  if (b < BB) {
    float row[U1];
    const float* lab_b = lab + (size_t)b * TT * UU;
    const float* bl_b  = blank + (size_t)b * TT * U1;
    row[0] = 0.0f;
    for (int u = 1; u <= UU; ++u) row[u] = row[u - 1] + lab_b[u - 1];
    int xl = xlen[b], yl = ylen[b];
    if (xl == 1) ll = row[yl] + bl_b[yl];
    for (int t = 1; t < TT; ++t) {
      const float* bl = bl_b + (size_t)(t - 1) * U1;
      const float* lb = lab_b + (size_t)t * UU;
      row[0] = row[0] + bl[0];
      for (int u = 1; u <= UU; ++u) {
        float atop = row[u] + bl[u];
        row[u] = logaddexpf_(atop, row[u - 1] + lb[u - 1]);
      }
      if (t == xl - 1) ll = row[yl] + bl_b[(size_t)t * U1 + yl];
    }
  }
#pragma unroll
  for (int off = 16; off >= 1; off >>= 1) ll += __shfl_xor(ll, off, 32);
  if (threadIdx.x == 0) out[0] = -ll * (1.0f / (float)BB);
}

// ---------------------------------------------------------------------------
// Host-side orchestration
// ---------------------------------------------------------------------------
extern "C" void kernel_launch(void* const* d_in, const int* in_sizes, int n_in,
                              void* d_out, int out_size, void* d_ws, size_t ws_size,
                              hipStream_t stream) {
  const float* xs     = (const float*)d_in[0];
  const int*   ymat   = (const int*)d_in[1];
  const int*   ys     = (const int*)d_in[2];
  const int*   xlen   = (const int*)d_in[3];
  const int*   ylen   = (const int*)d_in[4];
  // d_in[5] = embed (unused: one-hot handled analytically)
  const float* eWih0  = (const float*)d_in[6];
  const float* eWhh0  = (const float*)d_in[7];
  const float* ebih0  = (const float*)d_in[8];
  const float* ebhh0  = (const float*)d_in[9];
  const float* eWih1  = (const float*)d_in[10];
  const float* eWhh1  = (const float*)d_in[11];
  const float* ebih1  = (const float*)d_in[12];
  const float* ebhh1  = (const float*)d_in[13];
  const float* encW   = (const float*)d_in[14];
  const float* encb   = (const float*)d_in[15];
  const float* dWih   = (const float*)d_in[16];
  const float* dWhh   = (const float*)d_in[17];
  const float* dbih   = (const float*)d_in[18];
  const float* dbhh   = (const float*)d_in[19];
  const float* decW   = (const float*)d_in[20];
  const float* decb   = (const float*)d_in[21];
  float* out = (float*)d_out;

  // --- workspace carve-up (256B aligned) ---
  char* base = (char*)d_ws;
  size_t off = 0;
  auto carve = [&](size_t bytes) -> char* {
    off = (off + 255) & ~(size_t)255;
    char* p = base + off;
    off += bytes;
    return p;
  };
  const size_t packWhhElems = 8ull * 64 * 512;   // K=256,N=1024
  const size_t packWih0Elems = 3ull * 64 * 512;  // K=80 (padded),N=1024
  const size_t packPrjElems = 8ull * 8 * 512;    // K=256,N=128

  __bf16* pWih0 = (__bf16*)carve(packWih0Elems * 2);
  __bf16* pWhh0 = (__bf16*)carve(packWhhElems * 2);
  __bf16* pWih1 = (__bf16*)carve(packWhhElems * 2);
  __bf16* pWhh1 = (__bf16*)carve(packWhhElems * 2);
  __bf16* pWhhD = (__bf16*)carve(packWhhElems * 2);
  __bf16* pEncW = (__bf16*)carve(packPrjElems * 2);
  __bf16* pDecW = (__bf16*)carve(packPrjElems * 2);

  float* X0    = (float*)carve((size_t)BB * TT * GG * 4);  // reused for X1
  float* h0seq = (float*)carve((size_t)BB * TT * HH * 4);
  float* h1seq = (float*)carve((size_t)BB * TT * HH * 4);
  float* encL  = (float*)carve((size_t)BB * TT * VV * 4);
  float* Xd    = (float*)carve((size_t)BB * U1 * GG * 4);
  float* hdseq = (float*)carve((size_t)BB * U1 * HH * 4);
  float* decL  = (float*)carve((size_t)BB * U1 * VV * 4);
  float* blank = (float*)carve((size_t)BB * TT * U1 * 4);
  float* lab   = (float*)carve((size_t)BB * TT * UU * 4);
  (void)ws_size; (void)n_in; (void)in_sizes; (void)out_size;

  // --- 1) pack weights into bf16 WMMA fragment layout ---
  auto packLaunch = [&](const float* W, __bf16* p, int N, int K) {
    long total = (long)((K + 31) >> 5) * (N >> 4) * 512;
    pack_w<<<(unsigned)((total + 255) / 256), 256, 0, stream>>>(W, p, N, K);
  };
  packLaunch(eWih0, pWih0, GG, INP);
  packLaunch(eWhh0, pWhh0, GG, HH);
  packLaunch(eWih1, pWih1, GG, HH);
  packLaunch(eWhh1, pWhh1, GG, HH);
  packLaunch(dWhh,  pWhhD, GG, HH);
  packLaunch(encW,  pEncW, VV, HH);
  packLaunch(decW,  pDecW, VV, HH);

  // --- 2..5) encoder ---
  gemm_bf16<<<dim3(GG / 128, (BB * TT) / 16), 256, 0, stream>>>(
      xs, pWih0, ebih0, ebhh0, X0, BB * TT, GG, INP);
  lstm_seq<<<1, 512, 0, stream>>>(X0, pWhh0, h0seq, TT);
  gemm_bf16<<<dim3(GG / 128, (BB * TT) / 16), 256, 0, stream>>>(
      h0seq, pWih1, ebih1, ebhh1, X0, BB * TT, GG, HH);
  lstm_seq<<<1, 512, 0, stream>>>(X0, pWhh1, h1seq, TT);

  // --- 6) encoder projection ---
  gemm_bf16<<<dim3(VV / 128, (BB * TT) / 16), 256, 0, stream>>>(
      h1seq, pEncW, encb, nullptr, encL, BB * TT, VV, HH);

  // --- 7..9) decoder ---
  dec_gather<<<(BB * U1 * GG + 255) / 256, 256, 0, stream>>>(
      ymat, dWih, dbih, dbhh, Xd);
  lstm_seq<<<1, 512, 0, stream>>>(Xd, pWhhD, hdseq, U1);
  gemm_bf16<<<dim3(VV / 128, (BB * U1) / 16), 256, 0, stream>>>(
      hdseq, pDecW, decb, nullptr, decL, BB * U1, VV, HH);

  // --- 10) joint + log-softmax (streaming, lp never materialized) ---
  joint_lse<<<(BB * TT) / 8, 256, 0, stream>>>(encL, decL, ys, blank, lab);

  // --- 11) alpha recursion + loss ---
  alpha_loss<<<1, 32, 0, stream>>>(blank, lab, xlen, ylen, out);
}
